// AttentionWithDiversity_11244224381685
// MI455X (gfx1250) — compile-verified, matrix operand order checked by reading, not run on hardware
//
#include <hip/hip_runtime.h>
#include <hip/hip_bf16.h>

#define NN   8192
#define IDIM 512
#define ADIM 256

typedef __attribute__((ext_vector_type(16))) _Float16 v16h;
typedef __attribute__((ext_vector_type(8)))  _Float16 v8h;
typedef __attribute__((ext_vector_type(8)))  float    v8f;

typedef unsigned int tdm_u4 __attribute__((ext_vector_type(4)));
typedef int          tdm_i8 __attribute__((ext_vector_type(8)));
typedef int          tdm_i4 __attribute__((ext_vector_type(4)));

// LDS staging geometry (padded row strides for conflict-free ds_load_b128)
#define KROWB 528            // 512B K row + 16B pad (pad_interval=6, pad_amount=3)
#define VROWB 80             // 64B  V row + 16B pad (pad_interval=3, pad_amount=3)
#define KBYTES (32 * KROWB)  // 16896
#define VBYTES (256 * VROWB) // 20480
#define BUFB   (KBYTES + VBYTES)
#define SMEMB  (2 * BUFB)    // double buffered: 74752 B (<320KB WGP LDS)

// ---- helpers -------------------------------------------------------------

__device__ __forceinline__ v8f wmma16(v16h a, v16h b, v8f c) {
  return __builtin_amdgcn_wmma_f32_16x16x32_f16(false, a, false, b, (short)0, c,
                                                false, false);
}

// 16-half A/B operand: elems0-7 at p, elems8-15 at p+16 halves (+32B).
__device__ __forceinline__ v16h ld16(const _Float16* p) {
  v8h lo = *(const v8h*)(p);
  v8h hi = *(const v8h*)(p + 16);
  return __builtin_shufflevector(lo, hi, 0, 1, 2, 3, 4, 5, 6, 7,
                                 8, 9, 10, 11, 12, 13, 14, 15);
}

// exchange with lane^16 (ds_swizzle SWAPX16: xor=0x10, and=0x1f)
__device__ __forceinline__ float swapx16(float x) {
  int r = __builtin_amdgcn_ds_swizzle(__float_as_int(x), 0x401F);
  return __int_as_float(r);
}

// Issue one 2D TDM tile load (f16 elements) global -> LDS.
// td0/td1: tensor dims (elems), stride0: tensor_dim0_stride (elems),
// tile0/tile1: tile dims (elems), pad per D# encoding.
__device__ __forceinline__ void tdm_load_2d(unsigned lds_off,
                                            const _Float16* gptr,
                                            unsigned td0, unsigned td1,
                                            unsigned long long stride0,
                                            unsigned tile0, unsigned tile1,
                                            unsigned pad_int, unsigned pad_amt) {
  unsigned long long ga = (unsigned long long)(size_t)gptr;
  tdm_u4 g0;
  g0[0] = 1u;                                    // count=1, user descriptor
  g0[1] = lds_off;                               // LDS byte address
  g0[2] = (unsigned)(ga & 0xffffffffu);          // global_addr[31:0]
  g0[3] = (unsigned)((ga >> 32) & 0x01ffffffu)   // global_addr[56:32]
        | (2u << 30);                            // type=2 ("image")
  tdm_i8 g1;
  g1[0] = (int)((1u << 16)                       // data_size=1 -> 2 bytes
        | (1u << 20)                             // pad_enable
        | (pad_int << 22) | (pad_amt << 25));
  g1[1] = (int)((td0 & 0xffffu) << 16);          // abar_addr=0 | tensor_dim0 lo
  g1[2] = (int)((td0 >> 16) | ((td1 & 0xffffu) << 16));
  g1[3] = (int)((td1 >> 16) | (tile0 << 16));    // tensor_dim1 hi | tile_dim0
  g1[4] = (int)(tile1 & 0xffffu);                // tile_dim1 | tile_dim2=0
  g1[5] = (int)(unsigned)(stride0 & 0xffffffffu);
  g1[6] = (int)(unsigned)((stride0 >> 32) & 0xffffu);
  g1[7] = 0;
  tdm_i4 g2 = {1, 1, 0, 0};                      // tensor_dim2=1, tensor_dim3=1
  tdm_i4 g3 = {0, 0, 0, 0};
#if defined(__clang_major__) && (__clang_major__ >= 23)
  tdm_i8 g4 = {0, 0, 0, 0, 0, 0, 0, 0};
  __builtin_amdgcn_tensor_load_to_lds(g0, g1, g2, g3, g4, 0);
#else
  __builtin_amdgcn_tensor_load_to_lds(g0, g1, g2, g3, 0);
#endif
}

// ---- kernel 0: zero the scalar loss slot ---------------------------------

__global__ void init_loss_kernel(float* __restrict__ out) {
  if (threadIdx.x == 0) out[(size_t)NN * ADIM] = 0.0f;
}

// ---- kernel 1: QKV projection (f32 in -> f16 q,k + f16 v-transposed) -----

__global__ void qkv_proj_kernel(const float* __restrict__ x,
                                const float* __restrict__ Wq, const float* __restrict__ bq,
                                const float* __restrict__ Wk, const float* __restrict__ bk,
                                const float* __restrict__ Wv, const float* __restrict__ bv,
                                _Float16* __restrict__ qh,
                                _Float16* __restrict__ kh,
                                _Float16* __restrict__ vT) {
  const int tile = blockIdx.x * 4 + (threadIdx.x >> 5);
  const int per_mat = (NN / 16) * (ADIM / 16);
  const int mat = tile / per_mat;            // 0=q 1=k 2=v
  const int rem = tile % per_mat;
  const int mt  = rem / (ADIM / 16);
  const int nt  = rem % (ADIM / 16);

  const float* W    = (mat == 0) ? Wq : (mat == 1) ? Wk : Wv;
  const float* bias = (mat == 0) ? bq : (mat == 1) ? bk : bv;

  const int L = threadIdx.x & 31;
  const int lane16 = L & 15;
  const int half   = L >> 4;

  const int m = mt * 16 + lane16;
  const int n = nt * 16 + lane16;

  v8f c = {};
#pragma unroll
  for (int kc = 0; kc < IDIM / 32; ++kc) {
    const int kb = kc * 32 + half * 8;
    const float* xa = x + (size_t)m * IDIM + kb;
    const float* wb = W + (size_t)n * IDIM + kb;
    v16h a, b;
#pragma unroll
    for (int e = 0; e < 8; ++e) {
      a[e]     = (_Float16)xa[e];
      a[e + 8] = (_Float16)xa[e + 16];
      b[e]     = (_Float16)wb[e];
      b[e + 8] = (_Float16)wb[e + 16];
    }
    c = wmma16(a, b, c);
  }

  const float bb    = bias[n];
  const float scale = (mat == 0) ? 0.0625f : 1.0f;   // fold 1/sqrt(256) into q

#pragma unroll
  for (int r = 0; r < 8; ++r) {
    const float val = (c[r] + bb) * scale;
    const int mg = mt * 16 + half * 8 + r;
    if (mat == 2) {
      vT[(size_t)n * NN + mg] = (_Float16)val;
    } else {
      _Float16* dst = (mat == 0) ? qh : kh;
      dst[(size_t)mg * ADIM + n] = (_Float16)val;
    }
  }
}

// ---- kernel 2: flash attention + double-softmax entropy ------------------
// 4 waves/WG, each owns 16 query rows; K/V tiles staged in LDS by the TDM
// (each wave DMAs a quarter of the tile), double-buffered on TENSORcnt.

__global__ void attn_kernel(const _Float16* __restrict__ qh,
                            const _Float16* __restrict__ kh,
                            const _Float16* __restrict__ vT,
                            float* __restrict__ out) {
  __shared__ __align__(16) char smem[SMEMB];
  const unsigned smem_lo = (unsigned)(size_t)(&smem[0]);   // LDS byte offset

  const int wv = threadIdx.x >> 5;           // wave in workgroup (0..3)
  const int mt = blockIdx.x * 4 + wv;
  const int L = threadIdx.x & 31;
  const int lane16 = L & 15;
  const int half   = L >> 4;
  const int i = mt * 16 + lane16;            // query row owned by this lane

  // Q rows as B-operands (fixed for whole kernel)
  v16h qB[8];
#pragma unroll
  for (int kc = 0; kc < 8; ++kc)
    qB[kc] = ld16(qh + (size_t)i * ADIM + kc * 32 + half * 8);

  // Per-wave TDM issue: quarter tiles. K: 8 rows, V: 64 rows.
  auto issueK = [&](int buf, int jb) {
    tdm_load_2d(smem_lo + buf * BUFB + (unsigned)(wv * 8) * KROWB,
                kh + (size_t)(jb + wv * 8) * ADIM,
                ADIM, NN, ADIM, ADIM, 8, 6u, 3u);
  };
  auto issueV = [&](int buf, int jb) {
    tdm_load_2d(smem_lo + buf * BUFB + KBYTES + (unsigned)(wv * 64) * VROWB,
                vT + (size_t)(wv * 64) * NN + jb,
                NN, ADIM, NN, 32, 64, 3u, 3u);
  };

  v8f o[16];
#pragma unroll
  for (int t = 0; t < 16; ++t) o[t] = (v8f){};
  float m_run = -1.0e30f;
  float zsum  = 0.0f;

  const int T = NN / 32;

  // ---------------- pass A: online softmax + O accumulation --------------
  issueK(0, 0);  issueV(0, 0);
  issueK(1, 32); issueV(1, 32);
  for (int t = 0; t < T; ++t) {
    const int jb = t * 32;
    __builtin_amdgcn_s_wait_tensorcnt(2);    // this wave's current buffer done
    __syncthreads();                         // everyone's quarter landed
    const char* kbuf = smem + (t & 1) * BUFB;
    const char* vbuf = kbuf + KBYTES;

    v8f c0 = {}, c1 = {};
#pragma unroll
    for (int kc = 0; kc < 8; ++kc) {
      v16h kA = ld16((const _Float16*)(kbuf + lane16 * KROWB + kc * 64 + half * 16));
      c0 = wmma16(kA, qB[kc], c0);
    }
#pragma unroll
    for (int kc = 0; kc < 8; ++kc) {
      v16h kA = ld16((const _Float16*)(kbuf + (16 + lane16) * KROWB + kc * 64 + half * 16));
      c1 = wmma16(kA, qB[kc], c1);
    }

    float bm = c0[0];
#pragma unroll
    for (int r = 1; r < 8; ++r) bm = fmaxf(bm, c0[r]);
#pragma unroll
    for (int r = 0; r < 8; ++r) bm = fmaxf(bm, c1[r]);
    bm = fmaxf(bm, swapx16(bm));
    const float m_new = fmaxf(m_run, bm);
    const float corr  = __expf(m_run - m_new);

    float esum = 0.0f;
    v16h sB;
#pragma unroll
    for (int r = 0; r < 8; ++r) {
      const float e0 = __expf(c0[r] - m_new);
      const float e1 = __expf(c1[r] - m_new);
      esum += e0 + e1;
      sB[r]     = (_Float16)e0;
      sB[r + 8] = (_Float16)e1;
    }
    zsum = zsum * corr + esum;

#pragma unroll
    for (int nt = 0; nt < 16; ++nt) {
      v16h vA = ld16((const _Float16*)(vbuf + (nt * 16 + lane16) * VROWB + half * 16));
#pragma unroll
      for (int r = 0; r < 8; ++r) o[nt][r] *= corr;
      o[nt] = wmma16(vA, sB, o[nt]);
    }
    m_run = m_new;

    __syncthreads();                         // all waves done reading buffer
    const int jn = ((t + 2) & (T - 1)) * 32; // wrapped tail issues are benign
    issueK(t & 1, jn);
    issueV(t & 1, jn);
  }
  __builtin_amdgcn_s_wait_tensorcnt(0);
  __syncthreads();

  const float Z    = zsum + swapx16(zsum);
  const float invZ = 1.0f / Z;

#pragma unroll
  for (int nt = 0; nt < 16; ++nt) {
    float* dst = out + (size_t)i * ADIM + nt * 16 + half * 8;
    float4 w0 = make_float4(o[nt][0] * invZ, o[nt][1] * invZ,
                            o[nt][2] * invZ, o[nt][3] * invZ);
    float4 w1 = make_float4(o[nt][4] * invZ, o[nt][5] * invZ,
                            o[nt][6] * invZ, o[nt][7] * invZ);
    *(float4*)(dst)     = w0;
    *(float4*)(dst + 4) = w1;
  }

  // ---------------- pass B: entropy of softmax(scores) -------------------
  // scores s = exp(l-m)/Z; prob = exp(s)/S2.
  // -(sum p log(p+1e-6)) ~= log S2 - U/S2 - N*1e-6, U = sum e*s.
  const float mF = m_run;
  float s2 = 0.0f, us = 0.0f;
  issueK(0, 0);
  issueK(1, 32);
  for (int t = 0; t < T; ++t) {
    __builtin_amdgcn_s_wait_tensorcnt(1);
    __syncthreads();
    const char* kbuf = smem + (t & 1) * BUFB;

    v8f c0 = {}, c1 = {};
#pragma unroll
    for (int kc = 0; kc < 8; ++kc) {
      v16h kA = ld16((const _Float16*)(kbuf + lane16 * KROWB + kc * 64 + half * 16));
      c0 = wmma16(kA, qB[kc], c0);
    }
#pragma unroll
    for (int kc = 0; kc < 8; ++kc) {
      v16h kA = ld16((const _Float16*)(kbuf + (16 + lane16) * KROWB + kc * 64 + half * 16));
      c1 = wmma16(kA, qB[kc], c1);
    }
#pragma unroll
    for (int r = 0; r < 8; ++r) {
      float s = __expf(c0[r] - mF) * invZ;
      float e = __expf(s);
      s2 += e; us += e * s;
      s = __expf(c1[r] - mF) * invZ;
      e = __expf(s);
      s2 += e; us += e * s;
    }

    __syncthreads();
    issueK(t & 1, ((t + 2) & (T - 1)) * 32);
  }
  __builtin_amdgcn_s_wait_tensorcnt(0);

  s2 += swapx16(s2);
  us += swapx16(us);
  if (half == 0) {
    const float loss_row = __logf(s2) - us / s2 - 1.0e-6f * (float)NN;
    atomicAdd(out + (size_t)NN * ADIM, loss_row * (1.0f / (float)NN));
  }
}

// ---- launch --------------------------------------------------------------

extern "C" void kernel_launch(void* const* d_in, const int* in_sizes, int n_in,
                              void* d_out, int out_size, void* d_ws, size_t ws_size,
                              hipStream_t stream) {
  (void)in_sizes; (void)n_in; (void)out_size; (void)ws_size;
  const float* x  = (const float*)d_in[0];
  const float* Wq = (const float*)d_in[1];
  const float* bq = (const float*)d_in[2];
  const float* Wk = (const float*)d_in[3];
  const float* bk = (const float*)d_in[4];
  const float* Wv = (const float*)d_in[5];
  const float* bv = (const float*)d_in[6];
  float* out = (float*)d_out;

  // workspace: qh[N*A] f16 | kh[N*A] f16 | vT[A*N] f16  (12 MB total)
  _Float16* ws = (_Float16*)d_ws;
  _Float16* qh = ws;
  _Float16* kh = ws + (size_t)NN * ADIM;
  _Float16* vT = ws + 2 * (size_t)NN * ADIM;

  hipLaunchKernelGGL(init_loss_kernel, dim3(1), dim3(32), 0, stream, out);

  const int qkv_tiles = 3 * (NN / 16) * (ADIM / 16);
  hipLaunchKernelGGL(qkv_proj_kernel, dim3(qkv_tiles / 4), dim3(128), 0, stream,
                     x, Wq, bq, Wk, bk, Wv, bv, qh, kh, vT);

  hipLaunchKernelGGL(attn_kernel, dim3((NN / 16) / 4), dim3(128), 0, stream,
                     qh, kh, vT, out);
}
